// MultiHeadedAttention_52269751992977
// MI455X (gfx1250) — matrix operand order repaired and checked by reading, not compiled, not run on hardware
//
#include <hip/hip_runtime.h>
#include <hip/hip_bf16.h>

// ---------------------------------------------------------------------------
// MI455X (gfx1250) fused T5 multi-head attention.
// B=4, S=2048, D=512, H=8, DH=64. bf16 WMMA (v_wmma_f32_16x16x32_bf16),
// flash-attention style online softmax, f32 accumulation. Loops are kept
// free of loop-carried fragment buffers (they cost v_mov storms); instead the
// GEMM K-loop is fully unrolled so the compiler hoists load clauses across
// steps, and in attention the V/mask loads are issued before the softmax
// VALU section which hides their latency.
// ---------------------------------------------------------------------------

typedef __attribute__((ext_vector_type(16))) __bf16 v16bf;
typedef __attribute__((ext_vector_type(8)))  __bf16 v8bf;
typedef __attribute__((ext_vector_type(8)))  float  v8f;

constexpr int Bc  = 4;
constexpr int Sc  = 2048;
constexpr int Dc  = 512;
constexpr int Hc  = 8;
constexpr int DHc = 64;

// ------------------------------- helpers -----------------------------------

__device__ __forceinline__ unsigned short f2bf(float f) {
    union { __bf16 h; unsigned short u; } cv;
    cv.h = (__bf16)f;
    return cv.u;
}

// 16 contiguous bf16 (32 bytes) -> fragment (B-matrix: b[j] = B[16*half + j][n])
__device__ __forceinline__ v16bf ld_contig16(const unsigned short* p) {
    v8bf lo = *(const v8bf*)(p);
    v8bf hi = *(const v8bf*)(p + 8);
    return __builtin_shufflevector(lo, hi, 0,1,2,3,4,5,6,7,8,9,10,11,12,13,14,15);
}

// A-matrix 16x32 bf16 fragment: lane holds row m, K contiguous:
// half=0 -> K {0..15}; half=1 -> K {16..31}
__device__ __forceinline__ v16bf ld_afrag(const unsigned short* row, int hf) {
    v8bf lo = *(const v8bf*)(row + hf * 16);
    v8bf hi = *(const v8bf*)(row + hf * 16 + 8);
    return __builtin_shufflevector(lo, hi, 0,1,2,3,4,5,6,7,8,9,10,11,12,13,14,15);
}

__device__ __forceinline__ v8f wmma_bf16(v16bf a, v16bf b, v8f c) {
    return __builtin_amdgcn_wmma_f32_16x16x32_bf16(
        /*neg_a=*/false, a, /*neg_b=*/false, b,
        /*c_mod=*/(short)0, c, /*reuse_a=*/false, /*reuse_b=*/false);
}

// T5 bidirectional bucket: num_buckets=32 -> nb=16, max_exact=8, max_dist=128
// large = 8 + int(log2(a/8) / log2(16) * 8) = 8 + int(2*log2(a/8))
__device__ __forceinline__ int t5_bucket(int rel) {
    int rb = (rel > 0) ? 16 : 0;
    int a  = (rel < 0) ? -rel : rel;
    int idx;
    if (a < 8) {
        idx = a;
    } else {
        float lg = __log2f((float)a * 0.125f);
        idx = 8 + (int)(lg * 2.0f);
        idx = idx > 15 ? 15 : idx;
    }
    return rb + idx;
}

// --------------------------- conversion kernels ----------------------------

// query|key|value f32 -> bf16, packed [3][B*S*D]
__global__ void convert_x_kernel(const float* __restrict__ q,
                                 const float* __restrict__ k,
                                 const float* __restrict__ v,
                                 unsigned short* __restrict__ out) {
    size_t i  = (size_t)blockIdx.x * 256 + threadIdx.x;
    size_t N1 = (size_t)Bc * Sc * Dc;
    if (i < N1)            out[i] = f2bf(q[i]);
    else if (i < 2 * N1)   out[i] = f2bf(k[i - N1]);
    else if (i < 3 * N1)   out[i] = f2bf(v[i - 2 * N1]);
}

// W [K][N] f32 -> Wt [N][K] bf16 (optionally scaled: folds 1/sqrt(DH) into Wq)
__global__ void convert_w_kernel(const float* __restrict__ W,
                                 unsigned short* __restrict__ Wt,
                                 float scale) {
    int i = blockIdx.x * 256 + threadIdx.x;
    if (i >= Dc * Dc) return;
    int k = i / Dc, n = i % Dc;
    Wt[(size_t)n * Dc + k] = f2bf(W[(size_t)k * Dc + n] * scale);
}

// ----------------------------- projection GEMM -----------------------------
// Y[M=8192][512] = X @ W + b.  4 waves/block, wave = 16 rows x 64 cols,
// K-loop step 32, 4 WMMA per step, fully unrolled (16 steps) so the
// scheduler issues load clauses several steps ahead with partial waits.
// Epilogue mode:
//   0: Q  bf16 [B,H,S,DH]   1: K bf16 [B,H,S,DH]
//   2: Vt bf16 [B,H,DH,S]   3: f32 row-major to d_out
__global__ __launch_bounds__(128)
void gemm_kernel(const unsigned short* __restrict__ X,
                 const unsigned short* __restrict__ Wt,
                 const float* __restrict__ bias, float bias_scale,
                 unsigned short* __restrict__ out_bf,
                 float* __restrict__ out_f, int mode) {
    const int lane = threadIdx.x & 31;
    const int wave = threadIdx.x >> 5;
    const int n    = lane & 15;
    const int hf   = lane >> 4;
    const int M    = Bc * Sc;

    const int bm = blockIdx.x % (M / 64);
    const int bn = blockIdx.x / (M / 64);
    const int m0 = bm * 64 + wave * 16;
    const int n0 = bn * 64;

    v8f acc[4];
#pragma unroll
    for (int t = 0; t < 4; ++t) acc[t] = (v8f)(0.0f);

    const unsigned short* xr = X + (size_t)(m0 + n) * Dc;

#pragma unroll
    for (int kk = 0; kk < Dc; kk += 32) {
        v16bf a = ld_afrag(xr + kk, hf);
#pragma unroll
        for (int t = 0; t < 4; ++t) {
            v16bf bw = ld_contig16(Wt + (size_t)(n0 + t * 16 + n) * Dc + kk + hf * 16);
            acc[t] = wmma_bf16(a, bw, acc[t]);
        }
    }

#pragma unroll
    for (int t = 0; t < 4; ++t) {
        int col = n0 + t * 16 + n;
        float bv = bias[col] * bias_scale;
#pragma unroll
        for (int r = 0; r < 8; ++r) {
            int   row = m0 + hf * 8 + r;
            float val = acc[t][r] + bv;
            int b = row / Sc, s = row % Sc;
            int h = col / DHc, dh = col % DHc;
            if (mode == 0 || mode == 1) {
                out_bf[((size_t)((b * Hc + h) * Sc + s)) * DHc + dh] = f2bf(val);
            } else if (mode == 2) {
                out_bf[((size_t)((b * Hc + h) * DHc + dh)) * Sc + s] = f2bf(val);
            } else {
                out_f[(size_t)row * Dc + col] = val;
            }
        }
    }
}

// --------------------------- fused flash attention -------------------------
// Block = 4 waves, one (b,h); each wave owns 16 query rows, streams key
// blocks of 32 (two 16x16 score tiles), online softmax in f32, P restaged
// through LDS into A-fragment layout, P@V via WMMA into 4 f32 accumulators.
// V fragments and mask bytes are issued right after the score WMMAs so the
// softmax VALU section hides their latency before the P@V WMMAs.
__global__ __launch_bounds__(128)
void attn_kernel(const unsigned short* __restrict__ Q,
                 const unsigned short* __restrict__ K,
                 const unsigned short* __restrict__ Vt,
                 const float* __restrict__ rel_bias,
                 const unsigned char* __restrict__ mask,
                 unsigned short* __restrict__ ctx) {
    __shared__ float          sBias[32];
    __shared__ unsigned short sP[4][16 * 32];

    const int lane = threadIdx.x & 31;
    const int wave = threadIdx.x >> 5;
    const int n    = lane & 15;
    const int hf   = lane >> 4;

    const int qblk = blockIdx.x % (Sc / 64);
    const int bh   = blockIdx.x / (Sc / 64);
    const int h    = bh % Hc;
    const int b    = bh / Hc;
    const int q0   = qblk * 64 + wave * 16;

    if (threadIdx.x < 32) sBias[threadIdx.x] = rel_bias[threadIdx.x * Hc + h];
    __syncthreads();

    const unsigned short* Qb = Q  + (size_t)bh * Sc * DHc;
    const unsigned short* Kb = K  + (size_t)bh * Sc * DHc;
    const unsigned short* Vb = Vt + (size_t)bh * DHc * Sc;

    const v16bf aq0 = ld_afrag(Qb + (size_t)(q0 + n) * DHc + 0,  hf);
    const v16bf aq1 = ld_afrag(Qb + (size_t)(q0 + n) * DHc + 32, hf);

    v8f acc[4];
#pragma unroll
    for (int t = 0; t < 4; ++t) acc[t] = (v8f)(0.0f);
    float mst[8], lst[8];
#pragma unroll
    for (int r = 0; r < 8; ++r) { mst[r] = -3.0e38f; lst[r] = 0.0f; }

    unsigned short* myP = sP[wave];
    const unsigned char* mrow0 = mask + (size_t)(b * Sc + q0 + hf * 8) * Sc;

    for (int kb = 0; kb < Sc; kb += 32) {
        // ---- K fragments + scores: two 16x16 tiles, each 2 WMMA over DH ----
        v8f s0 = (v8f)(0.0f), s1 = (v8f)(0.0f);
        {
            v16bf b00 = ld_contig16(Kb + (size_t)(kb + n) * DHc + hf * 16);
            v16bf b01 = ld_contig16(Kb + (size_t)(kb + n) * DHc + 32 + hf * 16);
            v16bf b10 = ld_contig16(Kb + (size_t)(kb + 16 + n) * DHc + hf * 16);
            v16bf b11 = ld_contig16(Kb + (size_t)(kb + 16 + n) * DHc + 32 + hf * 16);
            s0 = wmma_bf16(aq0, b00, s0);
            s0 = wmma_bf16(aq1, b01, s0);
            s1 = wmma_bf16(aq0, b10, s1);
            s1 = wmma_bf16(aq1, b11, s1);
        }

        // ---- issue V fragments for this block (consumed after softmax) ----
        v16bf bv[4];
#pragma unroll
        for (int t = 0; t < 4; ++t)
            bv[t] = ld_contig16(Vb + (size_t)(t * 16 + n) * Sc + kb + hf * 16);

        // ---- mask bytes for this block (one early clause) ----
        unsigned char mb0[8], mb1[8];
#pragma unroll
        for (int r = 0; r < 8; ++r) {
            const unsigned char* mp = mrow0 + (size_t)r * Sc;
            mb0[r] = mp[kb + n];
            mb1[r] = mp[kb + 16 + n];
        }

        // ---- prefetch next block's K/V/mask cachelines ----
        const int nb = kb + 32;
        if (nb < Sc) {
            __builtin_prefetch(Kb + (size_t)(nb + n) * DHc, 0, 3);
            __builtin_prefetch(Vb + (size_t)n * Sc + nb + hf * 16, 0, 3);
            __builtin_prefetch(mrow0 + nb, 0, 3);
        }

        // ---- bias + mask + online softmax (row = q0 + hf*8 + r) ----
#pragma unroll
        for (int r = 0; r < 8; ++r) {
            const int qrow = q0 + hf * 8 + r;
            const int key0 = kb + n;
            const int key1 = kb + 16 + n;
            float v0 = s0[r] + sBias[t5_bucket(key0 - qrow)];
            float v1 = s1[r] + sBias[t5_bucket(key1 - qrow)];
            if (mb0[r]) v0 = -1.0e18f;
            if (mb1[r]) v1 = -1.0e18f;

            float mx = fmaxf(v0, v1);
#pragma unroll
            for (int d = 1; d < 16; d <<= 1) mx = fmaxf(mx, __shfl_xor(mx, d, 32));
            const float nm = fmaxf(mst[r], mx);

            const float p0 = __expf(v0 - nm);
            const float p1 = __expf(v1 - nm);
            float rs = p0 + p1;
#pragma unroll
            for (int d = 1; d < 16; d <<= 1) rs += __shfl_xor(rs, d, 32);

            const float alpha = __expf(mst[r] - nm);
            lst[r] = lst[r] * alpha + rs;
            mst[r] = nm;
#pragma unroll
            for (int t = 0; t < 4; ++t) acc[t][r] *= alpha;

            myP[(hf * 8 + r) * 32 + n]      = f2bf(p0);
            myP[(hf * 8 + r) * 32 + 16 + n] = f2bf(p1);
        }

        // ---- P @ V : one 16x32 A-frag from LDS, 4 dh tiles ----
        v16bf ap = ld_afrag(myP + n * 32, hf);
#pragma unroll
        for (int t = 0; t < 4; ++t) acc[t] = wmma_bf16(ap, bv[t], acc[t]);
    }

    // ---- epilogue: normalize, store ctx bf16 [B,S,D] ----
#pragma unroll
    for (int t = 0; t < 4; ++t) {
#pragma unroll
        for (int r = 0; r < 8; ++r) {
            const int   qrow = q0 + hf * 8 + r;
            const int   col  = h * DHc + t * 16 + n;
            const float val  = acc[t][r] / lst[r];
            ctx[(size_t)(b * Sc + qrow) * Dc + col] = f2bf(val);
        }
    }
}

// ------------------------------- launcher ----------------------------------

extern "C" void kernel_launch(void* const* d_in, const int* in_sizes, int n_in,
                              void* d_out, int out_size, void* d_ws, size_t ws_size,
                              hipStream_t stream) {
    const float* key_in   = (const float*)d_in[0];
    const float* value_in = (const float*)d_in[1];
    const float* query_in = (const float*)d_in[2];
    const float* Wq = (const float*)d_in[3];
    const float* bq = (const float*)d_in[4];
    const float* Wk = (const float*)d_in[5];
    const float* bk = (const float*)d_in[6];
    const float* Wv = (const float*)d_in[7];
    const float* bv = (const float*)d_in[8];
    const float* Wo = (const float*)d_in[9];
    const float* bo = (const float*)d_in[10];
    const float* rel_bias = (const float*)d_in[11];
    const unsigned char* mask = (const unsigned char*)d_in[12];
    float* out = (float*)d_out;

    const size_t N1 = (size_t)Bc * Sc * Dc;  // 4,194,304 elems
    const size_t DD = (size_t)Dc * Dc;       // 262,144 elems

    unsigned short* ws  = (unsigned short*)d_ws;
    unsigned short* Xbf = ws;                 // [3][N1] : query, key, value (bf16)
    unsigned short* Wqt = Xbf + 3 * N1;
    unsigned short* Wkt = Wqt + DD;
    unsigned short* Wvt = Wkt + DD;
    unsigned short* Wot = Wvt + DD;
    unsigned short* Qb  = Wot + DD;           // [B,H,S,DH]
    unsigned short* Kb  = Qb + N1;            // [B,H,S,DH]
    unsigned short* Vtb = Kb + N1;            // [B,H,DH,S]
    unsigned short* Ctx = Xbf;                // reuse X region after projections

    const int M = Bc * Sc;

    // 1) convert activations (query, key, value) to bf16
    convert_x_kernel<<<(unsigned)((3 * N1 + 255) / 256), 256, 0, stream>>>(
        query_in, key_in, value_in, Xbf);

    // 2) convert+transpose weights (fold 1/sqrt(DH)=0.125 into Wq)
    convert_w_kernel<<<(unsigned)((DD + 255) / 256), 256, 0, stream>>>(Wq, Wqt, 0.125f);
    convert_w_kernel<<<(unsigned)((DD + 255) / 256), 256, 0, stream>>>(Wk, Wkt, 1.0f);
    convert_w_kernel<<<(unsigned)((DD + 255) / 256), 256, 0, stream>>>(Wv, Wvt, 1.0f);
    convert_w_kernel<<<(unsigned)((DD + 255) / 256), 256, 0, stream>>>(Wo, Wot, 1.0f);

    // 3) Q/K/V projections (WMMA bf16)
    const unsigned gemm_grid = (unsigned)((M / 64) * (Dc / 64));  // 1024
    gemm_kernel<<<gemm_grid, 128, 0, stream>>>(Xbf + 0 * N1, Wqt, bq, 0.125f, Qb,  nullptr, 0);
    gemm_kernel<<<gemm_grid, 128, 0, stream>>>(Xbf + 1 * N1, Wkt, bk, 1.0f,   Kb,  nullptr, 1);
    gemm_kernel<<<gemm_grid, 128, 0, stream>>>(Xbf + 2 * N1, Wvt, bv, 1.0f,   Vtb, nullptr, 2);

    // 4) fused flash attention with T5 relative bias + mask
    const unsigned attn_grid = (unsigned)(Bc * Hc * (Sc / 64));   // 1024
    attn_kernel<<<attn_grid, 128, 0, stream>>>(Qb, Kb, Vtb, rel_bias, mask, Ctx);

    // 5) output projection -> f32 d_out
    gemm_kernel<<<gemm_grid, 128, 0, stream>>>(Ctx, Wot, bo, 1.0f, nullptr, out, 3);
}
